// AsymmetricProjectedLinear_27384711479892
// MI455X (gfx1250) — matrix operand order repaired and checked by reading, not compiled, not run on hardware
//
#include <hip/hip_runtime.h>
#include <stdint.h>

// CDNA5 / gfx1250: wave32, v_wmma_f32_16x16x32_bf16 (f32 accumulate).
typedef __attribute__((ext_vector_type(16))) __bf16 v16bf;
typedef __attribute__((ext_vector_type(8)))  float  v8f;
typedef __attribute__((ext_vector_type(4)))  int    v4i;

// ---------------------------------------------------------------------------
// Packed-fragment format: each 16x32 (A) / 32x16 (B) tile stored in a
// per-lane WMMA fragment layout (contiguous k = 16*half + i, not the ISA table order) as bf16 hi/lo pairs:
//    tile = 2048 B : [ lane0..31 hi v16bf ] (1024 B) [ lane0..31 lo ] (1024 B)
// A tiles indexed (mt*KT + kt), B tiles indexed (nt*KT + kt). In the GEMMs a
// lane loads only its own 32-byte v16bf (2x b128) — no hot-loop conversions.
// ---------------------------------------------------------------------------
static constexpr int TILE_BYTES = 2048;

__device__ __forceinline__ void split_store(const float f[16], char* dst) {
    v16bf hi, lo;
#pragma unroll
    for (int i = 0; i < 16; ++i) {
        const __bf16 h = (__bf16)f[i];
        hi[i] = h;
        lo[i] = (__bf16)(f[i] - (float)h);
    }
    *(v16bf*)dst          = hi;
    *(v16bf*)(dst + 1024) = lo;
}

// Pack A[M,K] (f32, lda) -> fragment tiles. grid=(KT, MT), block=32.
__global__ __launch_bounds__(32)
void pack_a(const float* __restrict__ A, char* __restrict__ Apk, int lda, int KT)
{
    const int lane = threadIdx.x;
    const int kt = blockIdx.x, mt = blockIdx.y;
    const int half = lane >> 4, l15 = lane & 15;
    const float* src = A + (size_t)(mt * 16 + l15) * lda + kt * 32 + half * 16;
    float f[16];
#pragma unroll
    for (int i = 0; i < 8; ++i) f[i]     = src[i];       // K = base + 0..7
#pragma unroll
    for (int i = 0; i < 8; ++i) f[8 + i] = src[8 + i];   // K = base + 8..15
    split_store(f, Apk + ((size_t)mt * KT + kt) * TILE_BYTES + lane * 32);
}

// Pack op(B)[K,N] -> fragment tiles. grid=(KT, NT), block=32.
template <bool TRANSB>
__global__ __launch_bounds__(32)
void pack_b(const float* __restrict__ B, char* __restrict__ Bpk, int ldb, int KT)
{
    const int lane = threadIdx.x;
    const int kt = blockIdx.x, nt = blockIdx.y;
    const int half = lane >> 4, l15 = lane & 15;
    const int n = nt * 16 + l15;
    const int kbase = kt * 32 + half * 16;
    float f[16];
#pragma unroll
    for (int i = 0; i < 16; ++i)
        f[i] = TRANSB ? B[(size_t)n * ldb + kbase + i]
                      : B[(size_t)(kbase + i) * ldb + n];
    split_store(f, Bpk + ((size_t)nt * KT + kt) * TILE_BYTES + lane * 32);
}

__device__ __forceinline__ v8f wmma3(v8f acc, v16bf ahi, v16bf alo,
                                     v16bf bhi, v16bf blo) {
    acc = __builtin_amdgcn_wmma_f32_16x16x32_bf16(false, ahi, false, bhi,
                                                  (short)0, acc, false, false);
    acc = __builtin_amdgcn_wmma_f32_16x16x32_bf16(false, ahi, false, blo,
                                                  (short)0, acc, false, false);
    acc = __builtin_amdgcn_wmma_f32_16x16x32_bf16(false, alo, false, bhi,
                                                  (short)0, acc, false, false);
    return acc;
}

// ---------------------------------------------------------------------------
// Single-wave GEMM (skinny N=64, split-K): wave computes a 16x16 tile.
// ---------------------------------------------------------------------------
__global__ __launch_bounds__(32)
void gemm_skinny(const char* __restrict__ Apk, const char* __restrict__ Bpk,
                 float* __restrict__ C, unsigned N, int KT,
                 int ktPerSlice, size_t sliceStride)
{
    const int lane = threadIdx.x;
    const int nt = blockIdx.x, mt = blockIdx.y;
    const int ktBeg = blockIdx.z * ktPerSlice;
    int ktEnd = ktBeg + ktPerSlice;
    if (ktEnd > KT) ktEnd = KT;
    float* Cs = C + (size_t)blockIdx.z * sliceStride;

    v8f acc = {};
    const char* aBase = Apk + (size_t)mt * KT * TILE_BYTES + lane * 32;
    const char* bBase = Bpk + (size_t)nt * KT * TILE_BYTES + lane * 32;

    for (int kt = ktBeg; kt < ktEnd; ++kt) {
        const char* at = aBase + (size_t)kt * TILE_BYTES;
        const char* bt = bBase + (size_t)kt * TILE_BYTES;
        // Branchless, in-bounds prefetch of next K-step's A tile (near caches).
        const int ktn = (kt + 1 < ktEnd) ? kt + 1 : kt;
        __builtin_prefetch(aBase + (size_t)ktn * TILE_BYTES, 0, 3);
        const v16bf ahi = *(const v16bf*)at;
        const v16bf alo = *(const v16bf*)(at + 1024);
        const v16bf bhi = *(const v16bf*)bt;
        const v16bf blo = *(const v16bf*)(bt + 1024);
        acc = wmma3(acc, ahi, alo, bhi, blo);
    }

    const unsigned half = lane >> 4, l15 = lane & 15;
#pragma unroll
    for (int r = 0; r < 8; ++r)
        Cs[(unsigned)(mt * 16 + r + half * 8) * N + (unsigned)(nt * 16) + l15] = acc[r];
}

// ---------------------------------------------------------------------------
// 4-wave LDS-staged GEMM: block = 4 waves sharing one ntg (4 N-tiles = 8 KB
// B panel per K-step, staged once into LDS, double-buffered), each wave owns
// one mt row of A. Async-to-LDS (ASYNCcnt) when the toolchain exposes it.
// ---------------------------------------------------------------------------
#if __has_builtin(__builtin_amdgcn_global_load_async_to_lds_b128)
#define STAGE_ASYNC 1
typedef __attribute__((address_space(1))) v4i* gv4ip;   // global int4*
typedef __attribute__((address_space(3))) v4i* lv4ip;   // LDS int4*
#else
#define STAGE_ASYNC 0
#endif

__global__ __launch_bounds__(128)
void gemm_lds(const char* __restrict__ Apk, const char* __restrict__ Bpk,
              float* __restrict__ C, unsigned N, int KT)
{
    constexpr int JT = 4;                       // N-tiles per block
    constexpr int WAVES = 4;                    // mt rows per block
    constexpr int PANEL = JT * TILE_BYTES;      // 8 KB per K-step
    __shared__ char sB[2 * PANEL];              // 16 KB double buffer

    const int tid  = threadIdx.x;
    const int lane = tid & 31;
    const int wv   = tid >> 5;
    const int ntg  = blockIdx.x;
    const int mt   = blockIdx.y * WAVES + wv;

    const char* aBase = Apk + (size_t)mt * KT * TILE_BYTES + lane * 32;
    const char* bBase = Bpk + (size_t)ntg * JT * KT * TILE_BYTES;

    v8f acc[JT] = {};

    // Stage B panel for K-step kt into LDS buffer `buf`:
    // thread t moves 16 B of tile j at intra-tile offset t*16 (128*16 = 2048).
    auto stage = [&](int kt, int buf) {
        const char* g = bBase + (size_t)kt * TILE_BYTES + (size_t)tid * 16;
        char* s = sB + buf * PANEL + tid * 16;
#pragma unroll
        for (int j = 0; j < JT; ++j) {
#if STAGE_ASYNC
            __builtin_amdgcn_global_load_async_to_lds_b128(
                (gv4ip)(uintptr_t)(g + (size_t)j * KT * TILE_BYTES),
                (lv4ip)(uint32_t)(uintptr_t)(s + j * TILE_BYTES), 0, 0);
#else
            *(float4*)(s + j * TILE_BYTES) =
                *(const float4*)(g + (size_t)j * KT * TILE_BYTES);
#endif
        }
    };

    stage(0, 0);
    for (int kt = 0; kt < KT; ++kt) {
        const int cur = kt & 1;
        const bool more = (kt + 1) < KT;
        if (more) stage(kt + 1, cur ^ 1);       // prefetch next panel
#if STAGE_ASYNC
        if (more) asm volatile("s_wait_asynccnt 4" ::: "memory");  // kt's 4 done
        else      asm volatile("s_wait_asynccnt 0" ::: "memory");
#endif
        __syncthreads();                         // panel visible to all waves

        const char* at = aBase + (size_t)kt * TILE_BYTES;
        const v16bf ahi = *(const v16bf*)at;
        const v16bf alo = *(const v16bf*)(at + 1024);
        const char* sb = sB + cur * PANEL + lane * 32;
#pragma unroll
        for (int j = 0; j < JT; ++j) {
            const v16bf bhi = *(const v16bf*)(sb + j * TILE_BYTES);
            const v16bf blo = *(const v16bf*)(sb + j * TILE_BYTES + 1024);
            acc[j] = wmma3(acc[j], ahi, alo, bhi, blo);
        }
        __syncthreads();                         // done reading before overwrite
    }

    const unsigned half = lane >> 4, l15 = lane & 15;
#pragma unroll
    for (int j = 0; j < JT; ++j)
#pragma unroll
        for (int r = 0; r < 8; ++r)
            C[(unsigned)(mt * 16 + r + half * 8) * N
              + (unsigned)((ntg * JT + j) * 16) + l15] = acc[j][r];
}

// Deterministic split-K reduction: dst[i] = sum_s part[s*MN + i].
__global__ __launch_bounds__(256)
void reduce_partials(const float* __restrict__ part, float* __restrict__ dst,
                     int S, size_t MN)
{
    const size_t i = (size_t)blockIdx.x * blockDim.x + threadIdx.x;
    if (i < MN) {
        float s = 0.f;
        for (int k = 0; k < S; ++k) s += part[(size_t)k * MN + i];
        dst[i] = s;
    }
}

// ---------------------------------------------------------------------------
// out = ((((x A_in) B_in) W_small^T) B_out^T) A_out^T      (~7 G MACs)
// ---------------------------------------------------------------------------
extern "C" void kernel_launch(void* const* d_in, const int* in_sizes, int n_in,
                              void* d_out, int out_size, void* d_ws, size_t ws_size,
                              hipStream_t stream)
{
    (void)in_sizes; (void)n_in; (void)out_size; (void)ws_size;

    const float* x      = (const float*)d_in[0];  // [4096, 4096]
    const float* Wsmall = (const float*)d_in[1];  // [1024, 1024]
    const float* A_out  = (const float*)d_in[2];  // [4096, 64]
    const float* B_out  = (const float*)d_in[3];  // [64, 1024]
    const float* A_in   = (const float*)d_in[4];  // [4096, 64]
    const float* B_in   = (const float*)d_in[5];  // [64, 1024]
    float* out = (float*)d_out;                   // [4096, 4096]

    const int M = 4096, DL = 4096, DS = 1024, R = 64;
    const int MT = M / 16;                 // 256
    const int KT_DL = DL / 32;             // 128
    const int KT_DS = DS / 32;             // 32
    const int KT_R  = R / 32;              // 2
    const int S = 8;                       // split-K slices for skinny GEMMs

    // ---- workspace layout (lifetime-based reuse, 94 MB total) ----
    const size_t MB = 1u << 20;
    char* ws = (char*)d_ws;
    // Region A (0..64MB): Pa_x, reused after GEMM1:
    char*  Pa_x    = ws;                           // 64 MB   [G1]
    char*  Pa_v    = ws;                           // 16 MB   [G3]
    char*  Pb_ws   = ws + 16 * MB;                 //  4 MB   [G3]
    char*  Pa_w    = ws + 20 * MB;                 // 16 MB   [G4]
    char*  Pb_bout = ws + 36 * MB;                 // .25 MB  [G4]
    float* part4   = (float*)(ws + 37 * MB);       //  8 MB   [G4]
    char*  Pa_y    = ws + 45 * MB;                 //  1 MB   [G5]
    // Region B (64..80MB): v f32, reused for w f32:
    float* v_f32   = (float*)(ws + 64 * MB);
    float* w_f32   = (float*)(ws + 64 * MB);
    // Region C (80..88MB): GEMM1 partials:
    float* part1   = (float*)(ws + 80 * MB);
    // Region D (88..90MB): u f32, reused for y f32:
    float* u_f32   = (float*)(ws + 88 * MB);
    float* y_f32   = (float*)(ws + 88 * MB);
    // Region E (90..92MB): Pa_u + Pb_bin:
    char*  Pa_u    = ws + 90 * MB;
    char*  Pb_bin  = ws + 91 * MB;
    // Region F (92..94MB): Pb_ain + Pb_aout:
    char*  Pb_ain  = ws + 92 * MB;
    char*  Pb_aout = ws + 93 * MB;

    const dim3 wv(32), blk(128);

    // ---- pack constant inputs ----
    pack_a<<<dim3(KT_DL, MT), wv, 0, stream>>>(x, Pa_x, DL, KT_DL);
    pack_b<false><<<dim3(KT_DL, R / 16), wv, 0, stream>>>(A_in, Pb_ain, R, KT_DL);
    pack_b<false><<<dim3(KT_R, DS / 16), wv, 0, stream>>>(B_in, Pb_bin, DS, KT_R);
    pack_b<true ><<<dim3(KT_R, DL / 16), wv, 0, stream>>>(A_out, Pb_aout, R, KT_R);

    // ---- G1: u = x @ A_in   [M,R], K=DL  (split-K over 8 slices) ----
    gemm_skinny<<<dim3(R / 16, MT, S), wv, 0, stream>>>(
        Pa_x, Pb_ain, part1, R, KT_DL, KT_DL / S, (size_t)M * R);
    reduce_partials<<<dim3((unsigned)(((size_t)M * R + 255) / 256)), dim3(256), 0, stream>>>(
        part1, u_f32, S, (size_t)M * R);
    pack_a<<<dim3(KT_R, MT), wv, 0, stream>>>(u_f32, Pa_u, R, KT_R);

    // ---- G2: v = u @ B_in   [M,DS], K=R ----
    gemm_lds<<<dim3(DS / 64, MT / 4), blk, 0, stream>>>(Pa_u, Pb_bin, v_f32, DS, KT_R);
    pack_a<<<dim3(KT_DS, MT), wv, 0, stream>>>(v_f32, Pa_v, DS, KT_DS);
    pack_b<true><<<dim3(KT_DS, DS / 16), wv, 0, stream>>>(Wsmall, Pb_ws, DS, KT_DS);

    // ---- G3: w = v @ W_small^T   [M,DS], K=DS ----
    gemm_lds<<<dim3(DS / 64, MT / 4), blk, 0, stream>>>(Pa_v, Pb_ws, w_f32, DS, KT_DS);
    pack_a<<<dim3(KT_DS, MT), wv, 0, stream>>>(w_f32, Pa_w, DS, KT_DS);
    pack_b<true><<<dim3(KT_DS, R / 16), wv, 0, stream>>>(B_out, Pb_bout, DS, KT_DS);

    // ---- G4: y = w @ B_out^T   [M,R], K=DS  (split-K) ----
    gemm_skinny<<<dim3(R / 16, MT, S), wv, 0, stream>>>(
        Pa_w, Pb_bout, part4, R, KT_DS, KT_DS / S, (size_t)M * R);
    reduce_partials<<<dim3((unsigned)(((size_t)M * R + 255) / 256)), dim3(256), 0, stream>>>(
        part4, y_f32, S, (size_t)M * R);
    pack_a<<<dim3(KT_R, MT), wv, 0, stream>>>(y_f32, Pa_y, R, KT_R);

    // ---- G5: out = y @ A_out^T   [M,DL], K=R ----
    gemm_lds<<<dim3(DL / 64, MT / 4), blk, 0, stream>>>(Pa_y, Pb_aout, out, DL, KT_R);
}
